// MultiHeadSelfAttention_70394513981490
// MI455X (gfx1250) — compile-verified, matrix operand order checked by reading, not run on hardware
//
#include <hip/hip_runtime.h>

// MHSA forward for MI455X (gfx1250), bf16 WMMA pipeline, v2.
// T=4096, DIM=1024, H=16, HD=64. ~103 GFLOP, fully L2-resident (192MB),
// so compute-bound on the WMMA pipe. v2 register-blocks the GEMMs
// (32x64 per wave -> 1.5 loads/wmma instead of 4), hoists the Q tile in
// the attention score phase, and pads the LDS score rows (stride 4100)
// to kill the 16-way bank conflict in the PV A-operand reads.

#define T_LEN  4096
#define DIM_   1024
#define NHEADS 16
#define HD     64
#define SROW   4100   // padded LDS row stride (floats); 4100 % 64 == 4

typedef __attribute__((ext_vector_type(16))) __bf16 v16bf;
typedef __attribute__((ext_vector_type(8)))  float  v8f;

union BF16x8 { uint4 q; __bf16 h[8]; };
union F32x16 { float4 v[4]; float f[16]; };

__device__ __forceinline__ void load8bf(const __bf16* p, v16bf& v, int base) {
  BF16x8 t;
  t.q = *(const uint4*)p;           // one global_load_b128 (8 bf16)
#pragma unroll
  for (int i = 0; i < 8; ++i) v[base + i] = t.h[i];
}

// Schraudolph fast-exp, matching reference semantics:
// y = A*x + B; window-mask to 0; value-convert to i32; bitcast to f32.
__device__ __forceinline__ float fastexp_gist(float x) {
  float y = fmaf(12102203.17133801f, x, 1064986823.010288f);
  if (y < 8388608.0f || y > 2139095040.0f) y = 0.0f;
  return __int_as_float((int)y);
}

// Quake fast rsqrt (reference: SCALING = q_rsqrt(HEAD_DIM))
__device__ __forceinline__ float qrsqrt(float x) {
  int i = __float_as_int(x);
  i = 0x5f3759df - (i >> 1);
  float y = __int_as_float(i);
  return y * (1.5f - 0.5f * x * y * y);
}

// ---------------------------------------------------------------- f32 -> bf16
__global__ void cvt_bf16_kernel(const float* __restrict__ src,
                                __bf16* __restrict__ dst, int n) {
  int i = (blockIdx.x * blockDim.x + threadIdx.x) * 8;
  if (i >= n) return;
  float4 a = *(const float4*)(src + i);
  float4 b = *(const float4*)(src + i + 4);
  BF16x8 p;
  p.h[0] = (__bf16)a.x; p.h[1] = (__bf16)a.y;
  p.h[2] = (__bf16)a.z; p.h[3] = (__bf16)a.w;
  p.h[4] = (__bf16)b.x; p.h[5] = (__bf16)b.y;
  p.h[6] = (__bf16)b.z; p.h[7] = (__bf16)b.w;
  *(uint4*)(dst + i) = p.q;
}

// ------------------------------------------------------- generic bf16 GEMM
// y[m,n] = sum_k A[m,k] * W[n,k] (+ bias[n])   (torch Linear: x @ W.T + b)
// MODE 0: bf16 row-major out (Q, K)
// MODE 1: bf16 TRANSPOSED out[n*ldo + m] (V -> V^T)
// MODE 2: f32 row-major out (final projection)
// Wave tile: 32(M) x 64(N) = 2x4 accumulators; block = 4 waves as 2x2
// -> block tile 64(M) x 128(N). 12 b128 loads per 8 wmma.
template <int MODE>
__global__ void gemm_bf16_kernel(const __bf16* __restrict__ A,
                                 const __bf16* __restrict__ W,
                                 const float* __restrict__ bias,
                                 void* __restrict__ out,
                                 int M, int N, int K, int ldo) {
  const int lane = threadIdx.x & 31;
  const int wave = threadIdx.x >> 5;
  const int wm0  = blockIdx.x * 64 + (wave & 1) * 32;   // wave's first row
  const int wn0  = blockIdx.y * 128 + (wave >> 1) * 64; // wave's first col
  const int kb   = (lane >> 4) * 8;   // K half-split per ISA A/B layout

  const __bf16* ap[2];
  const __bf16* wp[4];
#pragma unroll
  for (int i = 0; i < 2; ++i)
    ap[i] = A + (size_t)(wm0 + i * 16 + (lane & 15)) * K;
#pragma unroll
  for (int j = 0; j < 4; ++j)
    wp[j] = W + (size_t)(wn0 + j * 16 + (lane & 15)) * K;

  v8f acc[2][4] = {};
  for (int k0 = 0; k0 < K; k0 += 32) {
    __builtin_prefetch(ap[0] + k0 + 256, 0, 1);   // global_prefetch_b8
    __builtin_prefetch(wp[0] + k0 + 256, 0, 1);
    v16bf a[2], b[4];
#pragma unroll
    for (int i = 0; i < 2; ++i) {
      load8bf(ap[i] + k0 + kb,      a[i], 0);
      load8bf(ap[i] + k0 + kb + 16, a[i], 8);
    }
#pragma unroll
    for (int j = 0; j < 4; ++j) {
      load8bf(wp[j] + k0 + kb,      b[j], 0);
      load8bf(wp[j] + k0 + kb + 16, b[j], 8);
    }
#pragma unroll
    for (int i = 0; i < 2; ++i)
#pragma unroll
      for (int j = 0; j < 4; ++j)
        acc[i][j] = __builtin_amdgcn_wmma_f32_16x16x32_bf16(
            false, a[i], false, b[j], (short)0, acc[i][j], false, false);
  }

#pragma unroll
  for (int j = 0; j < 4; ++j) {
    const int col = wn0 + j * 16 + (lane & 15);
    const float bv = bias ? bias[col] : 0.0f;
#pragma unroll
    for (int i = 0; i < 2; ++i) {
      const int mbase = wm0 + i * 16 + (lane >> 4) * 8; // C: VGPR r <-> M=r/r+8
      if (MODE == 0) {                 // bf16 row-major
        __bf16* o = (__bf16*)out;
#pragma unroll
        for (int r = 0; r < 8; ++r)
          o[(size_t)(mbase + r) * ldo + col] = (__bf16)(acc[i][j][r] + bv);
      } else if (MODE == 1) {          // bf16 transposed, contiguous in m
        __bf16* o = (__bf16*)out;
        BF16x8 p;
#pragma unroll
        for (int r = 0; r < 8; ++r) p.h[r] = (__bf16)(acc[i][j][r] + bv);
        *(uint4*)(o + (size_t)col * ldo + mbase) = p.q;
      } else {                         // f32 row-major
        float* o = (float*)out;
#pragma unroll
        for (int r = 0; r < 8; ++r)
          o[(size_t)(mbase + r) * ldo + col] = acc[i][j][r] + bv;
      }
    }
  }
}

// ------------------------------------------------------------- attention
// One workgroup = one (head, 16-row tile of T). 4 waves / 128 threads.
// LDS: scores[16][SROW] f32 (~257KB) + pmax[16][8] + psum[16][8] + rowsum[16].
#define SMEM_FLOATS (16 * SROW + 16 * 8 + 16 * 8 + 16)
#define SMEM_BYTES  (SMEM_FLOATS * 4)

__global__ void attn_kernel(const __bf16* __restrict__ Qb,
                            const __bf16* __restrict__ Kb,
                            const __bf16* __restrict__ Vt,   // [DIM][T]
                            __bf16* __restrict__ Ob) {
  extern __shared__ float smem[];
  float* pmax   = smem + 16 * SROW;       // [16][8]
  float* psum   = pmax + 16 * 8;          // [16][8]
  float* rowsum = psum + 16 * 8;          // [16]

  const int h    = blockIdx.y;
  const int m0   = blockIdx.x * 16;
  const int tid  = threadIdx.x;
  const int lane = tid & 31;
  const int wave = tid >> 5;
  const int kb   = (lane >> 4) * 8;
  const float scaling = qrsqrt((float)HD);

  // ---- S = Q K^T * scaling -> LDS. Wave w covers cols [w*1024, w*1024+1024)
  // Q tile is loop-invariant per wave: hoist into registers.
  const __bf16* Qrow = Qb + (size_t)(m0 + (lane & 15)) * DIM_ + h * HD;
  v16bf qa[2];
#pragma unroll
  for (int i = 0; i < 2; ++i) {
    load8bf(Qrow + i * 32 + kb,      qa[i], 0);
    load8bf(Qrow + i * 32 + kb + 16, qa[i], 8);
  }
  for (int nt = 0; nt < 64; ++nt) {
    const int n0 = wave * 1024 + nt * 16;
    const __bf16* Krow = Kb + (size_t)(n0 + (lane & 15)) * DIM_ + h * HD;
    v8f acc = {};
#pragma unroll
    for (int i = 0; i < 2; ++i) {
      v16bf b;
      load8bf(Krow + i * 32 + kb,      b, 0);
      load8bf(Krow + i * 32 + kb + 16, b, 8);
      acc = __builtin_amdgcn_wmma_f32_16x16x32_bf16(
          false, qa[i], false, b, (short)0, acc, false, false);
    }
    const int mloc = (lane >> 4) * 8;
    const int ncol = n0 + (lane & 15);
#pragma unroll
    for (int r = 0; r < 8; ++r)
      smem[(mloc + r) * SROW + ncol] = acc[r] * scaling;
  }
  __syncthreads();

  // ---- softmax (fastexp): thread -> (row = tid&15, chunk of 512 cols)
  const int r  = tid & 15;
  const int ch = tid >> 4;
  float* srow = smem + r * SROW + ch * 512;
  float lm = -3.402823466e38f;
  for (int i = 0; i < 512; ++i) lm = fmaxf(lm, srow[i]);
  pmax[r * 8 + ch] = lm;
  __syncthreads();

  float mrow = pmax[r * 8 + 0];
#pragma unroll
  for (int i = 1; i < 8; ++i) mrow = fmaxf(mrow, pmax[r * 8 + i]);
  float ls = 0.0f;
  for (int i = 0; i < 512; ++i) {
    float e = fastexp_gist(srow[i] - mrow);
    srow[i] = e;
    ls += e;
  }
  psum[r * 8 + ch] = ls;
  __syncthreads();

  if (tid < 16) {
    float s = 0.0f;
#pragma unroll
    for (int i = 0; i < 8; ++i) s += psum[tid * 8 + i];
    rowsum[tid] = s;
  }
  __syncthreads();

  // ---- O = P V. Wave w owns head-dim cols [w*16, w*16+16). K-loop over s.
  const int n0 = wave * 16;
  const __bf16* Vrow = Vt + (size_t)(h * HD + n0 + (lane & 15)) * T_LEN;
  v8f acc = {};
  for (int s0 = 0; s0 < T_LEN; s0 += 32) {
    __builtin_prefetch(Vrow + s0 + 256, 0, 1);
    // A = P tile from LDS (f32 -> bf16 on the fly), ISA A layout.
    // Padded SROW spreads the 16 rows over all 64 LDS banks.
    const float* p0 = smem + (lane & 15) * SROW + s0 + kb;
    F32x16 u;
    u.v[0] = *(const float4*)(p0);         // ds_load_b128
    u.v[1] = *(const float4*)(p0 + 4);
    u.v[2] = *(const float4*)(p0 + 16);
    u.v[3] = *(const float4*)(p0 + 20);
    v16bf a, b;
#pragma unroll
    for (int i = 0; i < 16; ++i) a[i] = (__bf16)u.f[i];
    load8bf(Vrow + s0 + kb,      b, 0);
    load8bf(Vrow + s0 + kb + 16, b, 8);
    acc = __builtin_amdgcn_wmma_f32_16x16x32_bf16(
        false, a, false, b, (short)0, acc, false, false);
  }
  // normalize by row sums in the epilogue, store O as bf16 row-major [T, DIM]
  const int ncol = h * HD + n0 + (lane & 15);
#pragma unroll
  for (int rr = 0; rr < 8; ++rr) {
    const int mloc = (lane >> 4) * 8 + rr;
    float o = acc[rr] / rowsum[mloc];
    Ob[(size_t)(m0 + mloc) * DIM_ + ncol] = (__bf16)o;
  }
}

// --------------------------------------------------------------- launcher
extern "C" void kernel_launch(void* const* d_in, const int* in_sizes, int n_in,
                              void* d_out, int out_size, void* d_ws, size_t ws_size,
                              hipStream_t stream) {
  const float* x  = (const float*)d_in[0];
  const float* Wq = (const float*)d_in[1];
  const float* bq = (const float*)d_in[2];
  const float* Wk = (const float*)d_in[3];
  const float* bk = (const float*)d_in[4];
  const float* Wv = (const float*)d_in[5];
  const float* bv = (const float*)d_in[6];
  const float* Wo = (const float*)d_in[7];
  // d_in[8] = tgt_len (fixed 4096)

  // Workspace layout (48 MB total, assumed <= ws_size):
  char* ws = (char*)d_ws;
  __bf16* xb  = (__bf16*)(ws + ((size_t)0  << 20));  // 8 MB  x bf16
  __bf16* wqb = (__bf16*)(ws + ((size_t)8  << 20));  // 2 MB
  __bf16* wkb = (__bf16*)(ws + ((size_t)10 << 20));  // 2 MB
  __bf16* wvb = (__bf16*)(ws + ((size_t)12 << 20));  // 2 MB
  __bf16* wob = (__bf16*)(ws + ((size_t)14 << 20));  // 2 MB
  __bf16* Qb  = (__bf16*)(ws + ((size_t)16 << 20));  // 8 MB
  __bf16* Kb  = (__bf16*)(ws + ((size_t)24 << 20));  // 8 MB
  __bf16* Vt  = (__bf16*)(ws + ((size_t)32 << 20));  // 8 MB  V^T [DIM][T]
  __bf16* Ob  = (__bf16*)(ws + ((size_t)40 << 20));  // 8 MB

  const int nx = T_LEN * DIM_;       // 4M
  const int nw = DIM_ * DIM_;        // 1M
  cvt_bf16_kernel<<<nx / 8 / 256, 256, 0, stream>>>(x,  xb,  nx);
  cvt_bf16_kernel<<<nw / 8 / 256, 256, 0, stream>>>(Wq, wqb, nw);
  cvt_bf16_kernel<<<nw / 8 / 256, 256, 0, stream>>>(Wk, wkb, nw);
  cvt_bf16_kernel<<<nw / 8 / 256, 256, 0, stream>>>(Wv, wvb, nw);
  cvt_bf16_kernel<<<nw / 8 / 256, 256, 0, stream>>>(Wo, wob, nw);

  dim3 ggrid(T_LEN / 64, DIM_ / 128);      // (64, 8), 64x128 block tiles
  dim3 gblk(128);
  // Q, K: bf16 row-major; V: bf16 transposed [DIM][T]
  gemm_bf16_kernel<0><<<ggrid, gblk, 0, stream>>>(xb, wqb, bq, (void*)Qb,
                                                  T_LEN, DIM_, DIM_, DIM_);
  gemm_bf16_kernel<0><<<ggrid, gblk, 0, stream>>>(xb, wkb, bk, (void*)Kb,
                                                  T_LEN, DIM_, DIM_, DIM_);
  gemm_bf16_kernel<1><<<ggrid, gblk, 0, stream>>>(xb, wvb, bv, (void*)Vt,
                                                  T_LEN, DIM_, DIM_, T_LEN);

  dim3 agrid(T_LEN / 16, NHEADS);          // (256, 16)
  attn_kernel<<<agrid, gblk, SMEM_BYTES, stream>>>(Qb, Kb, Vt, Ob);

  gemm_bf16_kernel<2><<<ggrid, gblk, 0, stream>>>(Ob, wob, nullptr, d_out,
                                                  T_LEN, DIM_, DIM_, DIM_);
}
